// Stylev2ALAEGeneratorBlock_52226802319411
// MI455X (gfx1250) — matrix operand order repaired and checked by reading, not compiled, not run on hardware
//
#include <hip/hip_runtime.h>
#include <stdint.h>

typedef __bf16  bf16;
typedef __bf16  v8bf  __attribute__((ext_vector_type(8)));
typedef __bf16  v16bf __attribute__((ext_vector_type(16)));
typedef float   v8f   __attribute__((ext_vector_type(8)));

#define BATCH 8
#define CH    512
#define HIN   32
#define WIN   32
#define HO    64
#define WO    64
#define PIX   (HO*WO)          // 4096
#define GAIN  1.41421356237f   // sqrt(2)
#define HE33  (1.0f/48.0f)     // GAIN/sqrt(512*9)
#define HE_A  0.044194173824159216f   // 1/sqrt(512)
#define HE_RGB 0.0013258252147247766f // 0.03/sqrt(512)

// LDS tiles (bf16, row pitch 80B to spread banks, 16B-aligned fragment reads)
#define A_PITCH 80
#define A_BYTES (128*A_PITCH)   // 10240  : 128 cout x 32 k
#define B_PITCH 80
#define B_BYTES (128*B_PITCH)   // 10240  : 128 px   x 32 k
#define BUF_BYTES (A_BYTES + B_BYTES)   // 20480, double buffered

// ---------------------------------------------------------------- style = conv1x1(w)
__global__ __launch_bounds__(256) void style_kernel(const float* __restrict__ w,
                                                    const float* __restrict__ Aw,
                                                    const float* __restrict__ Ab,
                                                    float* __restrict__ style) {
    int id = blockIdx.x * 256 + threadIdx.x;      // 8*512
    int b = id >> 9, o = id & 511;
    const float* wr = w + b * CH;
    const float* ar = Aw + o * CH;
    float s = 0.f;
    for (int d = 0; d < CH; ++d) s += wr[d] * ar[d];
    style[id] = s * HE_A + Ab[o];
}

// ---------------------------------------------------------------- bilinear 2x upsample (align_corners), fp32 -> bf16
__global__ __launch_bounds__(256) void upsample_kernel(const float* __restrict__ x,
                                                       bf16* __restrict__ xup) {
    int idx = blockIdx.x * 256 + threadIdx.x;     // 8*512*64*64
    int xo = idx & 63, yo = (idx >> 6) & 63, c = (idx >> 12) & 511, b = idx >> 21;
    const float S = 31.0f / 63.0f;
    float cx = xo * S, cy = yo * S;
    int ix0 = (int)cx, iy0 = (int)cy;
    float fx = cx - ix0, fy = cy - iy0;
    int ix1 = ix0 + 1 < 31 ? ix0 + 1 : 31;
    int iy1 = iy0 + 1 < 31 ? iy0 + 1 : 31;
    const float* xb = x + (b * CH + c) * (HIN * WIN);
    float v00 = xb[iy0 * WIN + ix0], v01 = xb[iy0 * WIN + ix1];
    float v10 = xb[iy1 * WIN + ix0], v11 = xb[iy1 * WIN + ix1];
    float v0 = v00 * (1.f - fy) + v10 * fy;
    float v1 = v01 * (1.f - fy) + v11 * fy;
    xup[idx] = (bf16)(v0 * (1.f - fx) + v1 * fx);
}

// ---------------------------------------------------------------- demod: rsqrt(sum (style*he*W)^2 + eps)
__global__ __launch_bounds__(256) void demod_kernel(const float* __restrict__ W,   // [Cin,Cout,3,3]
                                                    const float* __restrict__ style,
                                                    float* __restrict__ dmod) {
    __shared__ float red[256];
    int bc = blockIdx.x;                 // 8*512
    int b = bc >> 9, cout = bc & 511;
    int tid = threadIdx.x;
    float sum = 0.f;
    for (int cin = tid; cin < CH; cin += 256) {
        float s = style[b * CH + cin] * HE33;
        const float* wp = W + (cin * CH + cout) * 9;
        #pragma unroll
        for (int t = 0; t < 9; ++t) { float v = s * wp[t]; sum += v * v; }
    }
    red[tid] = sum;
    __syncthreads();
    for (int st = 128; st > 0; st >>= 1) {
        if (tid < st) red[tid] += red[tid + st];
        __syncthreads();
    }
    if (tid == 0) dmod[bc] = rsqrtf(red[0] + 1e-8f);
}

// ---------------------------------------------------------------- pack bf16 A: [b][tap][cout][cin], flip taps
__global__ __launch_bounds__(256) void pack_kernel(const float* __restrict__ W,
                                                   const float* __restrict__ style,
                                                   const float* __restrict__ dmod,
                                                   bf16* __restrict__ Abuf) {
    int idx = blockIdx.x * 256 + threadIdx.x;     // 8*9*512*512
    int cin  = idx & 511;
    int cout = (idx >> 9) & 511;
    int t    = (idx >> 18) % 9;
    int b    = idx / (9 * CH * CH);
    float v = GAIN * HE33 * style[b * CH + cin] * dmod[b * CH + cout]
            * W[((size_t)cin * CH + cout) * 9 + (8 - t)];
    Abuf[idx] = (bf16)v;
}

// ---------------------------------------------------------------- implicit-GEMM conv 3x3 via WMMA bf16 + fused epilogue
// grid (coutBlk=4, rowPair=32, b=8), block 256 (8 waves).
// WG tile: 128 couts x 128 pixels (two output rows). Double-buffered LDS.
template <bool STORE_BF16>
__global__ __launch_bounds__(256) void gemm_conv(const bf16*  __restrict__ xin,   // [b][c][64][64] bf16
                                                 const bf16*  __restrict__ Abuf,  // [b][t][cout][cin] bf16
                                                 const float* __restrict__ noise, // [b][4096]
                                                 const float* __restrict__ Sw, const float* __restrict__ Sb,
                                                 const float* __restrict__ Bw, const float* __restrict__ Bb,
                                                 const float* __restrict__ bias,
                                                 bf16*  __restrict__ outb,
                                                 float* __restrict__ outf) {
    __shared__ __align__(16) char lds[2 * BUF_BYTES];

    const int tid   = threadIdx.x;
    const int b     = blockIdx.z;
    const int y0    = blockIdx.y;       // row pair: rows 2*y0, 2*y0+1
    const int cout0 = blockIdx.x * 128;

    const int lane = tid & 31;
    const int wv   = tid >> 5;
    const int wm   = (wv & 3) * 32;     // cout sub-tile (0/32/64/96)
    const int wn   = (wv >> 2) * 64;    // pixel sub-tile (0/64)
    const int lm   = lane & 15;
    const int lh   = lane >> 4;

    // uniform base pointers, 32-bit offsets
    const uint64_t* Asrc = (const uint64_t*)(Abuf + (size_t)b * 9 * CH * CH);
    const uint16_t* Bsrc = (const uint16_t*)(xin + (size_t)b * CH * PIX);

    // per-thread invariant staging coordinates
    const int a_m0 = tid >> 3, a_kq = tid & 7;    // A: qword id = tid + s*256 -> m += 32/s
    const int b_kp0 = tid >> 7, b_n  = tid & 127; // B: dword id = tid + s*256 -> kp += 2/s
    const int b_ny = b_n >> 6, b_nx = b_n & 63;

    v8f acc[2][4] = {};

    auto stage = [&](int buf, int step) {
        char* AlsB = lds + buf * BUF_BYTES;
        char* BlsB = AlsB + A_BYTES;
        const int t  = step >> 4;          // tap 0..8
        const int cb = step & 15;          // cin block
        const int kh = t / 3, kw = t % 3;
        // ---- A tile: 128 x 32 bf16 [m][k]; qword granularity
        const int qbase = (t * CH + cout0) * 128 + cb * 8;   // qword offset (row = 128 qwords)
        #pragma unroll
        for (int s = 0; s < 4; ++s) {
            int m = a_m0 + s * 32;
            uint64_t v = Asrc[qbase + m * 128 + a_kq];
            *(uint64_t*)(AlsB + m * A_PITCH + a_kq * 8) = v;
        }
        // ---- B tile (im2col, 2 rows): 128 px x 32 k [n][k], dword-packed, branchless pad
        int sy = 2 * y0 + b_ny + kh - 1;
        int sx = b_nx + kw - 1;
        int syc = sy < 0 ? 0 : (sy > 63 ? 63 : sy);
        int sxc = sx < 0 ? 0 : (sx > 63 ? 63 : sx);
        uint32_t okm = ((sy == syc) && (sx == sxc)) ? 0xFFFFFFFFu : 0u;
        int off_n = syc * 64 + sxc;
        #pragma unroll
        for (int s = 0; s < 8; ++s) {
            int kp = b_kp0 + s * 2;
            int kbase = (cb * 32 + kp * 2) * PIX;            // uniform per s
            uint32_t lo = Bsrc[kbase + off_n];
            uint32_t hi = Bsrc[kbase + PIX + off_n];
            uint32_t v = (lo | (hi << 16)) & okm;
            *(uint32_t*)(BlsB + b_n * B_PITCH + kp * 4) = v;
        }
    };

    auto compute = [&](int buf) {
        const char* AlsB = lds + buf * BUF_BYTES;
        const char* BlsB = AlsB + A_BYTES;
        v16bf afr[2], bfr[4];
        #pragma unroll
        for (int i = 0; i < 2; ++i) {
            const char* pa = AlsB + (wm + i * 16 + lm) * A_PITCH + lh * 16;
            v8bf lo = *(const v8bf*)pa;
            v8bf h2 = *(const v8bf*)(pa + 32);
            afr[i] = __builtin_shufflevector(lo, h2, 0,1,2,3,4,5,6,7,8,9,10,11,12,13,14,15);
        }
        #pragma unroll
        for (int j = 0; j < 4; ++j) {
            const char* pb = BlsB + (wn + j * 16 + lm) * B_PITCH + lh * 16;
            v8bf lo = *(const v8bf*)pb;
            v8bf h2 = *(const v8bf*)(pb + 32);
            bfr[j] = __builtin_shufflevector(lo, h2, 0,1,2,3,4,5,6,7,8,9,10,11,12,13,14,15);
        }
        #pragma unroll
        for (int i = 0; i < 2; ++i)
            #pragma unroll
            for (int j = 0; j < 4; ++j)
                acc[i][j] = __builtin_amdgcn_wmma_f32_16x16x32_bf16(
                    false, afr[i], false, bfr[j], (short)0, acc[i][j], false, false);
    };

    stage(0, 0);
    __syncthreads();
    for (int s = 0; s < 144; ++s) {
        int cur = s & 1;
        int sn  = (s + 1 < 144) ? s + 1 : 143;   // last prefetch harmlessly redundant
        stage(cur ^ 1, sn);
        compute(cur);
        __syncthreads();
    }

    // ---- fused epilogue: x*ys + yb + bias, leaky_relu(0.2)
    float nv[4];
    int   pp[4];
    #pragma unroll
    for (int j = 0; j < 4; ++j) {
        int n  = wn + j * 16 + lm;              // 0..127
        int yy = 2 * y0 + (n >> 6);
        pp[j]  = yy * 64 + (n & 63);
        nv[j]  = noise[b * PIX + pp[j]];
    }
    #pragma unroll
    for (int i = 0; i < 2; ++i) {
        #pragma unroll
        for (int v = 0; v < 8; ++v) {
            int cout = cout0 + wm + i * 16 + lh * 8 + v;
            float sw = Sw[cout], sb = Sb[cout];
            float bw = Bw[cout], bb = Bb[cout];
            float bs = bias[cout];
            size_t obase = ((size_t)b * CH + cout) * PIX;
            #pragma unroll
            for (int j = 0; j < 4; ++j) {
                float o = acc[i][j][v] * (nv[j] * sw + sb) + (nv[j] * bw + bb) + bs;
                o = o > 0.f ? o : 0.2f * o;
                if (STORE_BF16) outb[obase + pp[j]] = (bf16)o;
                else            outf[obase + pp[j]] = o;
            }
        }
    }
}

// ---------------------------------------------------------------- toRGB 1x1 conv
__global__ __launch_bounds__(256) void rgb_kernel(const float* __restrict__ xf,   // [b][512][4096]
                                                  const float* __restrict__ rw,   // [3][512]
                                                  const float* __restrict__ rb,
                                                  float* __restrict__ out) {      // [b][3][4096]
    int id = blockIdx.x * 256 + threadIdx.x;     // 8*4096
    int b = id >> 12, p = id & 4095;
    const float* xb = xf + (size_t)b * CH * PIX + p;
    float r = 0.f, g = 0.f, bl = 0.f;
    for (int c = 0; c < CH; ++c) {
        float v = xb[(size_t)c * PIX];
        r  += v * rw[c];
        g  += v * rw[CH + c];
        bl += v * rw[2 * CH + c];
    }
    float* ob = out + (size_t)b * 3 * PIX + p;
    ob[0]       = r  * HE_RGB + rb[0];
    ob[PIX]     = g  * HE_RGB + rb[1];
    ob[2 * PIX] = bl * HE_RGB + rb[2];
}

// ----------------------------------------------------------------
extern "C" void kernel_launch(void* const* d_in, const int* in_sizes, int n_in,
                              void* d_out, int out_size, void* d_ws, size_t ws_size,
                              hipStream_t stream) {
    const float* x       = (const float*)d_in[0];
    const float* w       = (const float*)d_in[1];
    const float* noise1  = (const float*)d_in[2];
    const float* noise2  = (const float*)d_in[3];
    const float* conv1_w = (const float*)d_in[4];
    const float* bias1   = (const float*)d_in[5];
    const float* conv2_w = (const float*)d_in[6];
    const float* bias2   = (const float*)d_in[7];
    const float* A1w     = (const float*)d_in[8];
    const float* A1b     = (const float*)d_in[9];
    const float* A2w     = (const float*)d_in[10];
    const float* A2b     = (const float*)d_in[11];
    const float* B1sw    = (const float*)d_in[12];
    const float* B1sb    = (const float*)d_in[13];
    const float* B1bw    = (const float*)d_in[14];
    const float* B1bb    = (const float*)d_in[15];
    const float* B2sw    = (const float*)d_in[16];
    const float* B2sb    = (const float*)d_in[17];
    const float* B2bw    = (const float*)d_in[18];
    const float* B2bb    = (const float*)d_in[19];
    const float* rgbw    = (const float*)d_in[20];
    const float* rgbb    = (const float*)d_in[21];

    char* ws = (char*)d_ws;
    float* style1 = (float*)(ws);
    float* style2 = (float*)(ws + 16384);
    float* dmod   = (float*)(ws + 32768);
    bf16*  Abuf   = (bf16*)(ws + 49152);
    bf16*  xup    = (bf16*)(ws + 49152 + 37748736ull);
    bf16*  y1     = (bf16*)(ws + 49152 + 37748736ull + 33554432ull);

    float* outx   = (float*)d_out;
    float* outrgb = outx + (size_t)BATCH * CH * PIX;

    style_kernel<<<16, 256, 0, stream>>>(w, A1w, A1b, style1);
    style_kernel<<<16, 256, 0, stream>>>(w, A2w, A2b, style2);
    upsample_kernel<<<65536, 256, 0, stream>>>(x, xup);

    demod_kernel<<<4096, 256, 0, stream>>>(conv1_w, style1, dmod);
    pack_kernel<<<73728, 256, 0, stream>>>(conv1_w, style1, dmod, Abuf);
    gemm_conv<true><<<dim3(4, 32, 8), 256, 0, stream>>>(
        xup, Abuf, noise1, B1sw, B1sb, B1bw, B1bb, bias1, y1, nullptr);

    demod_kernel<<<4096, 256, 0, stream>>>(conv2_w, style2, dmod);
    pack_kernel<<<73728, 256, 0, stream>>>(conv2_w, style2, dmod, Abuf);
    gemm_conv<false><<<dim3(4, 32, 8), 256, 0, stream>>>(
        y1, Abuf, noise2, B2sw, B2sb, B2bw, B2bb, bias2, nullptr, outx);

    rgb_kernel<<<128, 256, 0, stream>>>(outx, rgbw, rgbb, outrgb);
}
